// MambaMixer_9835475108379
// MI455X (gfx1250) — matrix operand order repaired and checked by reading, not compiled, not run on hardware
//
#include <hip/hip_runtime.h>

// ---------------------------------------------------------------------------
// Mamba mixer forward on gfx1250 (MI455X).
//  * GEMMs: LDS-tiled bf16 WMMA (V_WMMA_F32_16X16X32_BF16, fp32 accumulate).
//    Block tile 128x128x32. A tile is DMA'd global->LDS with the gfx1250
//    async-to-LDS path (ASYNCcnt) and converted f32->bf16 at fragment-read
//    time (VALU co-executes with 16-bit WMMA); B tile is transposed+converted
//    through VGPRs into LDS. Each wave owns a 32x64 patch = eight 16x16
//    accumulators.
//  * Selective scan is recurrent -> fp32 VALU, lane-per-channel, state in
//    VGPRs, v_exp_f32 hidden by 128 concurrent wave32s.
// ---------------------------------------------------------------------------

typedef __attribute__((ext_vector_type(16))) __bf16 v16bf;
typedef __attribute__((ext_vector_type(8)))  __bf16 v8bf;
typedef __attribute__((ext_vector_type(8)))  float  v8f;

// Pointer/element type the async-to-LDS builtin expects (from the compiler
// diagnostic: "__attribute__((__vector_size__(4*sizeof(int)))) int *" in the
// generic address space).
typedef int async_v4i __attribute__((vector_size(16)));

#if __has_builtin(__builtin_amdgcn_global_load_async_to_lds_b128)
#define HAVE_ASYNC_LDS 1
#else
#define HAVE_ASYNC_LDS 0
#endif

__device__ __forceinline__ void wait_asynccnt0() {
#if HAVE_ASYNC_LDS
#if __has_builtin(__builtin_amdgcn_s_wait_asynccnt)
    __builtin_amdgcn_s_wait_asynccnt(0);
#else
    asm volatile("s_wait_asynccnt 0x0" ::: "memory");
#endif
#endif
}

#define D_MODEL 1024
#define D_STATE 16
#define D_CONV  4
#define D_INNER 2048
#define DT_RANK 64
#define BB      2
#define LL      1024
#define NTOK    (BB * LL)                 // 2048 tokens
#define NPARAM  (DT_RANK + 2 * D_STATE)   // 96

#define BM 128
#define BN 128
#define BK 32

__device__ __forceinline__ float sigmoidf_(float x) {
    return 1.0f / (1.0f + __expf(-x));
}

__device__ __forceinline__ v8bf pack8(float4 a, float4 b) {
    v8bf r;
    r[0] = (__bf16)a.x; r[1] = (__bf16)a.y; r[2] = (__bf16)a.z; r[3] = (__bf16)a.w;
    r[4] = (__bf16)b.x; r[5] = (__bf16)b.y; r[6] = (__bf16)b.z; r[7] = (__bf16)b.w;
    return r;
}

// ---------------------------------------------------------------------------
// LDS-tiled bf16 WMMA GEMM:  C[M,N] = epilogue(A[M,K(lda)] * B[K,N(ldb)])
// mode 0: plain store. mode 1: softplus(acc + bias[n]) (delta path).
// Block: 256 threads = 8 wave32, tiles BMxBN=128x128, k-step 32.
// Waves laid out 4(row) x 2(col); each wave: 2x4 WMMA accumulators (32x64).
// M multiple of 128, K multiple of 32 (true at all call sites); N guarded.
// ---------------------------------------------------------------------------
__global__ void __launch_bounds__(256)
wmma_gemm_bf16_f32(const float* __restrict__ A,
                   const float* __restrict__ B,
                   float* __restrict__ C,
                   const float* __restrict__ bias,
                   int M, int N, int K,
                   int lda, int ldb, int ldc,
                   int mode)
{
    __shared__ float  Asf[BM * BK];       // raw f32 A tile (async target), 16 KB
    __shared__ __bf16 Bs[BN * BK];        // transposed bf16 Bs[n][k], 8 KB

    const int tid  = threadIdx.x;
    const int lane = tid & 31;
    const int wave = tid >> 5;
    const int wrow = wave >> 1;           // 0..3  -> 32-row slab
    const int wcol = wave & 1;            // 0..1  -> 64-col slab
    const int lm   = lane & 15;           // fragment row/col within 16
    const int kh   = lane >> 4;           // half-wave K selector (ISA 7.12.2)

    const int m0 = blockIdx.y * BM;       // global row base of block
    const int n0 = blockIdx.x * BN;       // global col base of block

    // staging coordinates
    const int sa_r  = tid >> 1;           // A: row 0..127
    const int sa_cg = (tid & 1) * 16;     // A: k-group 0/16
    const int sb_k  = tid >> 3;           // B: k row 0..31
    const int sb_ng = (tid & 7) * 16;     // B: col group 0..112

    v8f acc[2][4] = {};

    for (int k0 = 0; k0 < K; k0 += BK) {
        // ---- stage A tile: async DMA global->LDS (raw f32, no VGPR trip) ----
        {
            const float* src = A + (size_t)(m0 + sa_r) * lda + k0 + sa_cg;
            float* dst = &Asf[sa_r * BK + sa_cg];
            if (k0 + BK < K) __builtin_prefetch(src + BK, 0, 1); // global_prefetch_b8
#if HAVE_ASYNC_LDS
#pragma unroll
            for (int q = 0; q < 4; ++q)
                __builtin_amdgcn_global_load_async_to_lds_b128(
                    (async_v4i*)(src + 4 * q), (async_v4i*)(dst + 4 * q), 0, 0);
#else
#pragma unroll
            for (int q = 0; q < 4; ++q)
                *(float4*)(dst + 4 * q) = *(const float4*)(src + 4 * q);
#endif
        }
        // ---- stage B tile transposed+converted: Bs[n][k] ----
        {
            const int gn = n0 + sb_ng;
            const float* src = B + (size_t)(k0 + sb_k) * ldb + gn;
            if (gn + 16 <= N) {
                float4 f0 = *(const float4*)(src);
                float4 f1 = *(const float4*)(src + 4);
                float4 f2 = *(const float4*)(src + 8);
                float4 f3 = *(const float4*)(src + 12);
                const float fv[16] = { f0.x,f0.y,f0.z,f0.w, f1.x,f1.y,f1.z,f1.w,
                                       f2.x,f2.y,f2.z,f2.w, f3.x,f3.y,f3.z,f3.w };
#pragma unroll
                for (int e = 0; e < 16; ++e)
                    Bs[(sb_ng + e) * BK + sb_k] = (__bf16)fv[e];
            } else {
#pragma unroll
                for (int e = 0; e < 16; ++e)
                    Bs[(sb_ng + e) * BK + sb_k] =
                        (gn + e < N) ? (__bf16)src[e] : (__bf16)0.0f;
            }
        }
        wait_asynccnt0();                 // this wave's DMA complete
        __syncthreads();                  // all waves' staging visible

        // ---- fragments from LDS ----
        // A: read f32 rows, convert to bf16 here (co-executes with WMMA).
        v16bf afrag[2], bfrag[4];
#pragma unroll
        for (int i = 0; i < 2; ++i) {
            const int m = wrow * 32 + i * 16 + lm;
            const float* ap = &Asf[m * BK];
            float4 q0 = *(const float4*)(ap + kh * 8);           // K=kh*8+0..3
            float4 q1 = *(const float4*)(ap + kh * 8 + 4);       // K=kh*8+4..7
            float4 q2 = *(const float4*)(ap + 16 + kh * 8);      // K=16+kh*8+0..3
            float4 q3 = *(const float4*)(ap + 16 + kh * 8 + 4);  // K=16+kh*8+4..7
            v8bf lo = pack8(q0, q1);
            v8bf hi = pack8(q2, q3);
            afrag[i] = __builtin_shufflevector(lo, hi,
                0,1,2,3,4,5,6,7,8,9,10,11,12,13,14,15);
        }
        // B: 16B bf16 chunks (K = kh*8 + 0..7, 16 + kh*8 + 0..7 at fixed column).
#pragma unroll
        for (int j = 0; j < 4; ++j) {
            const int n = wcol * 64 + j * 16 + lm;
            v8bf lo = *(const v8bf*)&Bs[n * BK + kh * 8];
            v8bf hi = *(const v8bf*)&Bs[n * BK + 16 + kh * 8];
            bfrag[j] = __builtin_shufflevector(lo, hi,
                0,1,2,3,4,5,6,7,8,9,10,11,12,13,14,15);
        }

        // ---- 2x4 WMMA on the matrix pipe ----
#pragma unroll
        for (int i = 0; i < 2; ++i)
#pragma unroll
            for (int j = 0; j < 4; ++j)
                acc[i][j] = __builtin_amdgcn_wmma_f32_16x16x32_bf16(
                    false, afrag[i], false, bfrag[j], (short)0, acc[i][j],
                    false, false);

        __syncthreads();
    }

    // ---- epilogue + store: lane owns col n, VGPR r -> row +r+kh*8 ----
#pragma unroll
    for (int j = 0; j < 4; ++j) {
        const int n = n0 + wcol * 64 + j * 16 + lm;
        if (n >= N) continue;
        const float bv = (mode == 1) ? bias[n] : 0.0f;
#pragma unroll
        for (int i = 0; i < 2; ++i) {
            const int mb = m0 + wrow * 32 + i * 16 + kh * 8;
#pragma unroll
            for (int r = 0; r < 8; ++r) {
                float v = acc[i][j][r];
                if (mode == 1) {          // fused bias + softplus
                    v += bv;
                    v = (v > 20.0f) ? v : log1pf(__expf(v));
                }
                C[(size_t)(mb + r) * ldc + n] = v;
            }
        }
    }
}

// ---------------------------------------------------------------------------
// Causal depthwise conv1d (kernel 4, left pad) + SiLU. xb = xz[:, :, :D_INNER].
// ---------------------------------------------------------------------------
__global__ void conv_silu_kernel(const float* __restrict__ xz,
                                 const float* __restrict__ conv_w,
                                 const float* __restrict__ conv_b,
                                 float* __restrict__ u)
{
    const int i = blockIdx.x * blockDim.x + threadIdx.x;
    if (i >= NTOK * D_INNER) return;
    const int d = i % D_INNER;
    const int t = (i / D_INNER) % LL;
    const int b = i / (D_INNER * LL);
    float acc = conv_b[d];
#pragma unroll
    for (int j = 0; j < D_CONV; ++j) {
        const int tt = t + j - (D_CONV - 1);
        if (tt >= 0)
            acc += xz[((size_t)(b * LL + tt)) * (2 * D_INNER) + d]
                   * conv_w[d * D_CONV + j];
    }
    u[i] = acc * sigmoidf_(acc);
}

// ---------------------------------------------------------------------------
// Selective scan: one lane per (b, d) channel; 16-wide state in VGPRs.
// ---------------------------------------------------------------------------
__global__ void ssm_scan_kernel(const float* __restrict__ delta,
                                const float* __restrict__ u,
                                const float* __restrict__ params,
                                const float* __restrict__ A_log,
                                float* __restrict__ ys)
{
    const int ch = blockIdx.x * blockDim.x + threadIdx.x;  // b*D_INNER + d
    if (ch >= BB * D_INNER) return;
    const int d = ch % D_INNER;
    const int b = ch / D_INNER;

    float a_s[D_STATE], st[D_STATE];
#pragma unroll
    for (int s = 0; s < D_STATE; ++s) {
        a_s[s] = -__expf(A_log[d * D_STATE + s]);
        st[s]  = 0.0f;
    }
    for (int t = 0; t < LL; ++t) {
        const size_t row = (size_t)b * LL + t;
        const float dl = delta[row * D_INNER + d];
        const float uu = u[row * D_INNER + d];
        const float du = dl * uu;
        const float* p = params + row * NPARAM;   // [dt(64) | B(16) | C(16)]
        float y = 0.0f;
#pragma unroll
        for (int s = 0; s < D_STATE; ++s) {
            const float da = __expf(dl * a_s[s]);
            const float v  = da * st[s] + du * p[DT_RANK + s];
            st[s] = v;
            y += v * p[DT_RANK + D_STATE + s];
        }
        ys[row * D_INNER + d] = y;
    }
}

// ---------------------------------------------------------------------------
// y = (ys + D*u) * silu(zb), in place on ys. zb = xz[:, :, D_INNER:].
// ---------------------------------------------------------------------------
__global__ void gate_kernel(const float* __restrict__ xz,
                            const float* __restrict__ u,
                            const float* __restrict__ Dp,
                            float* __restrict__ ys)
{
    const int i = blockIdx.x * blockDim.x + threadIdx.x;
    if (i >= NTOK * D_INNER) return;
    const int d = i % D_INNER;
    const size_t row = i / D_INNER;
    const float z = xz[row * (2 * D_INNER) + D_INNER + d];
    ys[i] = (ys[i] + Dp[d] * u[i]) * (z * sigmoidf_(z));
}

// ---------------------------------------------------------------------------
extern "C" void kernel_launch(void* const* d_in, const int* in_sizes, int n_in,
                              void* d_out, int out_size, void* d_ws, size_t ws_size,
                              hipStream_t stream)
{
    const float* x      = (const float*)d_in[0];
    const float* W_in   = (const float*)d_in[1];
    const float* conv_w = (const float*)d_in[2];
    const float* conv_b = (const float*)d_in[3];
    const float* W_x    = (const float*)d_in[4];
    const float* W_dt   = (const float*)d_in[5];
    const float* b_dt   = (const float*)d_in[6];
    const float* W_out  = (const float*)d_in[7];
    const float* A_log  = (const float*)d_in[8];
    const float* Dp     = (const float*)d_in[9];
    float* out = (float*)d_out;

    // Workspace layout (floats): xz(8M) | u(4M) | params(192K) | delta(4M) | ys(4M)
    float* ws     = (float*)d_ws;
    float* xz     = ws;
    float* u      = xz     + (size_t)NTOK * 2 * D_INNER;
    float* params = u      + (size_t)NTOK * D_INNER;
    float* delta  = params + (size_t)NTOK * NPARAM;
    float* ys     = delta  + (size_t)NTOK * D_INNER;

    const dim3 blk(256);

    // 1) xz = x @ W_in                 [2048,1024] x [1024,4096]
    wmma_gemm_bf16_f32<<<dim3((2 * D_INNER + BN - 1) / BN, NTOK / BM), blk, 0, stream>>>(
        x, W_in, xz, nullptr,
        NTOK, 2 * D_INNER, D_MODEL, D_MODEL, 2 * D_INNER, 2 * D_INNER, 0);

    // 2) u = silu(causal_conv(xb))
    conv_silu_kernel<<<(NTOK * D_INNER + 255) / 256, blk, 0, stream>>>(
        xz, conv_w, conv_b, u);

    // 3) params = u @ W_x              [2048,2048] x [2048,96]
    wmma_gemm_bf16_f32<<<dim3((NPARAM + BN - 1) / BN, NTOK / BM), blk, 0, stream>>>(
        u, W_x, params, nullptr,
        NTOK, NPARAM, D_INNER, D_INNER, NPARAM, NPARAM, 0);

    // 4) delta = softplus(params[:, :64] @ W_dt + b_dt)   [2048,64] x [64,2048]
    wmma_gemm_bf16_f32<<<dim3((D_INNER + BN - 1) / BN, NTOK / BM), blk, 0, stream>>>(
        params, W_dt, delta, b_dt,
        NTOK, D_INNER, DT_RANK, NPARAM, D_INNER, D_INNER, 1);

    // 5) selective scan over L
    ssm_scan_kernel<<<(BB * D_INNER + 255) / 256, blk, 0, stream>>>(
        delta, u, params, A_log, ys);

    // 6) gate: ys = (ys + D*u) * silu(zb)
    gate_kernel<<<(NTOK * D_INNER + 255) / 256, blk, 0, stream>>>(
        xz, u, Dp, ys);

    // 7) out = ys @ W_out              [2048,2048] x [2048,1024]
    wmma_gemm_bf16_f32<<<dim3((D_MODEL + BN - 1) / BN, NTOK / BM), blk, 0, stream>>>(
        ys, W_out, out, nullptr,
        NTOK, D_MODEL, D_INNER, D_INNER, D_MODEL, D_MODEL, 0);
}
